// PatchAggregator_f_34119220199673
// MI455X (gfx1250) — matrix operand order repaired and checked by reading, not compiled, not run on hardware
//
#include <hip/hip_runtime.h>
#include <hip/hip_bf16.h>
#include <math.h>

typedef __bf16 bf16_t;
typedef __attribute__((ext_vector_type(16))) __bf16 v16bf;
typedef __attribute__((ext_vector_type(8)))  float  v8f;
typedef __attribute__((ext_vector_type(4)))  unsigned int v4u;
typedef __attribute__((ext_vector_type(8)))  int v8i;
typedef __attribute__((ext_vector_type(4)))  int v4i;

#define DIM   512
#define QN    16
#define BL    4096   // B*L
#define NG    512    // N groups
#define GP    64     // P group size

// ---------------------------------------------------------------------------
// K0a: A_t[q][e] = dot(Wk[e,:], Q0[q,:])  stored TRANSPOSED in bf16 (16 x 512)
// so k_project's score B-fragment is a contiguous 32B load per lane.
// ---------------------------------------------------------------------------
__global__ __launch_bounds__(256) void k_proj_q(const float* __restrict__ Wk,
                                                const float* __restrict__ q0,
                                                bf16_t* __restrict__ A_t) {
  int t = blockIdx.x * 256 + threadIdx.x;     // 8192 threads
  int e = t >> 4, q = t & 15;
  const float* wr = Wk + e * DIM;
  const float* qr = q0 + q * DIM;
  float s = 0.f;
  for (int d = 0; d < DIM; ++d) s += wr[d] * qr[d];
  A_t[q * DIM + e] = (bf16_t)s;
}

// ---------------------------------------------------------------------------
// K0b: VQ[q][d] = Q0[q,:]·Wv[:,d] + bv[d]   (16 x 512, f32)
// ---------------------------------------------------------------------------
__global__ __launch_bounds__(256) void k_vq(const float* __restrict__ Wv,
                                            const float* __restrict__ q0,
                                            const float* __restrict__ bv,
                                            float* __restrict__ VQ) {
  int t = blockIdx.x * 256 + threadIdx.x;     // 8192 threads
  int q = t >> 9, d = t & 511;
  float s = bv[d];
  const float* qr = q0 + q * DIM;
  for (int e = 0; e < DIM; ++e) s += qr[e] * Wv[e * DIM + d];
  VQ[q * DIM + d] = s;
}

// ---------------------------------------------------------------------------
// K0c: s_self[q] = scale * Q0[q]·(Q0[q]@Wk + bk) ; bkq[q] = bk·Q0[q]
// ---------------------------------------------------------------------------
__global__ __launch_bounds__(256) void k_self(const float* __restrict__ Wk,
                                              const float* __restrict__ q0,
                                              const float* __restrict__ bk,
                                              float* __restrict__ sself,
                                              float* __restrict__ bkq) {
  const int q = blockIdx.x;
  const int t = threadIdx.x;
  __shared__ float red[256];
  __shared__ float redb[256];
  const float* qr = q0 + q * DIM;
  float acc = 0.f, accb = 0.f;
  for (int d = t; d < DIM; d += 256) {
    float kqd = 0.f;
    for (int e = 0; e < DIM; ++e) kqd += qr[e] * Wk[e * DIM + d];
    acc  += qr[d] * kqd;
    accb += bk[d] * qr[d];
  }
  red[t] = acc; redb[t] = accb;
  __syncthreads();
  for (int s = 128; s > 0; s >>= 1) {
    if (t < s) { red[t] += red[t + s]; redb[t] += redb[t + s]; }
    __syncthreads();
  }
  if (t == 0) {
    const float scale = 0.044194173824159216f;  // 1/sqrt(512)
    sself[q] = (red[0] + redb[0]) * scale;
    bkq[q]   = redb[0];
  }
}

// ---------------------------------------------------------------------------
// K0d: Wv_t[col][k] = bf16(Wv[k][col])  transposed bf16 copy (512 KB)
// ---------------------------------------------------------------------------
__global__ __launch_bounds__(256) void k_cvt_wv(const float* __restrict__ Wv,
                                                bf16_t* __restrict__ Wv_t) {
  int t = blockIdx.x * 256 + threadIdx.x;     // 262144 threads
  int col = t >> 9, k = t & 511;
  Wv_t[(size_t)col * DIM + k] = (bf16_t)Wv[(size_t)k * DIM + col];
}

// ---------------------------------------------------------------------------
// K1: Vp_bf16 = bf16(patch @ Wv + bv)  and  s_patch = scale*(patch@A + bkq)
// One block = 16 rows of patch_flat. 8 waves x 4 col-tiles of 16 = 512 cols.
// B fragments are single contiguous 32B v16bf loads from transposed operands.
// ---------------------------------------------------------------------------
__global__ __launch_bounds__(256) void k_project(const float* __restrict__ patch,
                                                 const bf16_t* __restrict__ Wv_t,
                                                 const float* __restrict__ bv,
                                                 const bf16_t* __restrict__ A_t,
                                                 const float* __restrict__ bkq,
                                                 bf16_t* __restrict__ Vp,
                                                 float* __restrict__ s_patch) {
  const int r0   = blockIdx.x * 16;
  const int wave = threadIdx.x >> 5;
  const int lane = threadIdx.x & 31;
  const int hi   = lane >> 4;           // half-wave select
  const int lm   = lane & 15;
  const int row  = r0 + lm;             // A-matrix row owned by this lane
  const int c0   = wave * 64;

  v8f acc[4] = {};
  v8f accs   = {};

  for (int kk = 0; kk < DIM; kk += 32) {
    // A fragment: 16x32 bf16, contiguous k = 16*hi + s order (vectorized loads)
    const float4* p4 = (const float4*)(patch + (size_t)row * DIM + kk + hi * 16);
    const float4 x0 = p4[0], x1 = p4[1];   // K = base+0..7
    const float4 x2 = p4[2], x3 = p4[3];   // K = base+8..15
    v16bf a;
    a[0]=(bf16_t)x0.x; a[1]=(bf16_t)x0.y; a[2]=(bf16_t)x0.z; a[3]=(bf16_t)x0.w;
    a[4]=(bf16_t)x1.x; a[5]=(bf16_t)x1.y; a[6]=(bf16_t)x1.z; a[7]=(bf16_t)x1.w;
    a[8]=(bf16_t)x2.x; a[9]=(bf16_t)x2.y; a[10]=(bf16_t)x2.z; a[11]=(bf16_t)x2.w;
    a[12]=(bf16_t)x3.x; a[13]=(bf16_t)x3.y; a[14]=(bf16_t)x3.z; a[15]=(bf16_t)x3.w;

    const int kB = kk + hi * 16;       // B fragment K base for this half-wave
#pragma unroll
    for (int t = 0; t < 4; ++t) {
      const int col = c0 + t * 16 + lm;
      const v16bf b = *(const v16bf*)(Wv_t + (size_t)col * DIM + kB);
      acc[t] = __builtin_amdgcn_wmma_f32_16x16x32_bf16(
          false, a, false, b, (short)0, acc[t], false, false);
    }
    if (wave == 0) {
      const v16bf bq = *(const v16bf*)(A_t + (size_t)lm * DIM + kB);
      accs = __builtin_amdgcn_wmma_f32_16x16x32_bf16(
          false, a, false, bq, (short)0, accs, false, false);
    }
  }

  // C/D layout: VGPR j -> M = j + hi*8, lane -> N = lm
#pragma unroll
  for (int t = 0; t < 4; ++t) {
    const int col = c0 + t * 16 + lm;
    const float bvc = bv[col];
#pragma unroll
    for (int j = 0; j < 8; ++j) {
      const int M = j + hi * 8;
      Vp[(size_t)(r0 + M) * DIM + col] = (bf16_t)(acc[t][j] + bvc);
    }
  }
  if (wave == 0) {
    const float scale = 0.044194173824159216f;
    const float bq = bkq[lm];
#pragma unroll
    for (int j = 0; j < 8; ++j) {
      const int M = j + hi * 8;
      s_patch[(size_t)(r0 + M) * QN + lm] = (accs[j] + bq) * scale;
    }
  }
}

// ---------------------------------------------------------------------------
// K3: per group n: gather scores, masked softmax over P+1, WMMA w@Vg,
// add self term, fusion attention pool over q, write out[n,:].
// Gathered V rows are staged into LDS with the Tensor Data Mover in
// gather mode (16 x 16-bit row indices per descriptor) when available.
// ---------------------------------------------------------------------------
__global__ __launch_bounds__(256) void k_aggregate(const int* __restrict__ indices,
                                                   const unsigned char* __restrict__ mask,
                                                   const float* __restrict__ s_patch,
                                                   const float* __restrict__ sself,
                                                   const float* __restrict__ VQ,
                                                   const float* __restrict__ Wf,
                                                   const float* __restrict__ bfp,
                                                   const bf16_t* __restrict__ Vp,
                                                   float* __restrict__ out) {
  __shared__ int    sIdx[GP];
  __shared__ float  sS[QN][GP];        // scores, overwritten in-place by weights
  __shared__ float  sW0[QN];
  __shared__ float  sFlog[QN];
  __shared__ float  sFw[QN];
  __shared__ bf16_t sVg[32][DIM];      // one K-stage of gathered V rows (32KB)
  __shared__ float  sOut[DIM];

  const int n   = blockIdx.x;
  const int tid = threadIdx.x;

  // gather indices + masked scores
  if (tid < GP) {
    const int ix = indices[n * GP + tid];
    sIdx[tid] = ix;
    const bool m = mask[n * GP + tid] != 0;
    const float* sp = s_patch + (size_t)ix * QN;
#pragma unroll
    for (int q = 0; q < QN; ++q) sS[q][tid] = m ? sp[q] : -3.0e38f;
  }
  if (tid < QN) sFlog[tid] = 0.f;
  for (int i = tid; i < DIM; i += 256) sOut[i] = 0.f;
  __syncthreads();

  // softmax over [self, P] per query (one thread per q)
  if (tid < QN) {
    const int q = tid;
    const float ss = sself[q];
    float m = ss;
    for (int p = 0; p < GP; ++p) m = fmaxf(m, sS[q][p]);
    float z = __expf(ss - m);
    const float w0 = z;
    for (int p = 0; p < GP; ++p) {
      const float e = __expf(sS[q][p] - m);
      sS[q][p] = e;
      z += e;
    }
    const float inv = 1.f / z;
    for (int p = 0; p < GP; ++p) sS[q][p] *= inv;
    sW0[q] = w0 * inv;
  }
  __syncthreads();

  const int wave = tid >> 5;
  const int lane = tid & 31;
  const int hi   = lane >> 4;
  const int lm   = lane & 15;
  const int c0   = wave * 64;

  v8f acc[4] = {};

  for (int stage = 0; stage < 2; ++stage) {
    const int pk = stage * 32;
    __syncthreads();   // protect sVg from overwrite while prior stage in use

#if __has_builtin(__builtin_amdgcn_tensor_load_to_lds) && \
    __has_builtin(__builtin_amdgcn_s_wait_tensorcnt)
    // --- TDM gather-mode staging: 2 descriptors x 16 gathered rows --------
    if (wave == 0) {
      const unsigned int ldsBase = (unsigned int)(size_t)(&sVg[0][0]);
      const unsigned long long ga = (unsigned long long)(size_t)Vp;
#pragma unroll
      for (int half = 0; half < 2; ++half) {
        v4u g0;
        g0[0] = 1u | (1u << 31);                       // count=1, gather_mode, 16b idx
        g0[1] = ldsBase + (unsigned int)half * 16u * (DIM * 2u);  // lds_addr
        g0[2] = (unsigned int)ga;                      // global_addr[31:0]
        g0[3] = (unsigned int)((ga >> 32) & 0x01FFFFFFu) | (2u << 30); // [56:32], type=2
        v8i g1;
        g1[0] = (int)(1u << 16);        // workgroup_mask=0, data_size=1 (2 bytes)
        g1[1] = (int)(512u << 16);      // atomic_barrier=0 | tensor_dim0[15:0]=512
        g1[2] = (int)(4096u << 16);     // tensor_dim0[31:16]=0 | tensor_dim1[15:0]=4096
        g1[3] = (int)(512u << 16);      // tensor_dim1[31:16]=0 | tile_dim0=512
        g1[4] = 16;                     // tile_dim1 = 16 valid indices
        g1[5] = 512;                    // tensor_dim0_stride[31:0] = 512
        g1[6] = 0;                      // stride hi / dim1_stride (ignored in gather)
        g1[7] = 0;
        const int base = pk + half * 16;
        v4i g2, g3;
#pragma unroll
        for (int j = 0; j < 4; ++j) {
          g2[j] = (sIdx[base + 2 * j] & 0xffff) | (sIdx[base + 2 * j + 1] << 16);
          g3[j] = (sIdx[base + 8 + 2 * j] & 0xffff) | (sIdx[base + 9 + 2 * j] << 16);
        }
        const v8i gz = {};   // trailing descriptor group (unused; zero-filled)
        __builtin_amdgcn_tensor_load_to_lds(g0, g1, g2, g3, gz, 0);
      }
      __builtin_amdgcn_s_wait_tensorcnt(0);
    }
#else
    // --- manual staging fallback: 8 threads per row, 128B each (uint4) ----
    {
      const int r  = tid >> 3;
      const int cs = (tid & 7) * 64;
      const uint4* src = (const uint4*)(Vp + (size_t)sIdx[pk + r] * DIM + cs);
      uint4* dst = (uint4*)(&sVg[r][cs]);
#pragma unroll
      for (int j = 0; j < 8; ++j) dst[j] = src[j];
    }
#endif
    __syncthreads();

    // A fragment: w (16 queries x 32 tokens), f32 -> bf16 from LDS
    v16bf a;
    const int pA = pk + hi * 16;
#pragma unroll
    for (int i = 0; i < 8; ++i) {
      a[i]     = (bf16_t)sS[lm][pA + i];
      a[i + 8] = (bf16_t)sS[lm][pA + 8 + i];
    }
    const int pB = hi * 16;   // local row base within this stage
#pragma unroll
    for (int t = 0; t < 4; ++t) {
      const int col = c0 + t * 16 + lm;
      v16bf b;
#pragma unroll
      for (int i = 0; i < 16; ++i) b[i] = sVg[pB + i][col];
      acc[t] = __builtin_amdgcn_wmma_f32_16x16x32_bf16(
          false, a, false, b, (short)0, acc[t], false, false);
    }
  }

  // patch_token values held per lane: M = j + hi*8, N = col
  float ptv[4][8];
#pragma unroll
  for (int t = 0; t < 4; ++t) {
    const int col = c0 + t * 16 + lm;
#pragma unroll
    for (int j = 0; j < 8; ++j) {
      const int M = j + hi * 8;
      ptv[t][j] = acc[t][j] + sW0[M] * VQ[M * DIM + col];
    }
  }

  // fusion logits: flog[q] = sum_d pt[q,d]*Wf[d]
#pragma unroll
  for (int j = 0; j < 8; ++j) {
    float s = 0.f;
#pragma unroll
    for (int t = 0; t < 4; ++t) s += ptv[t][j] * Wf[c0 + t * 16 + lm];
    atomicAdd(&sFlog[j + hi * 8], s);
  }
  __syncthreads();

  if (tid == 0) {
    const float b0 = bfp[0];
    float m = -3.0e38f;
    for (int q = 0; q < QN; ++q) m = fmaxf(m, sFlog[q] + b0);
    float z = 0.f;
    float e[QN];
    for (int q = 0; q < QN; ++q) { e[q] = __expf(sFlog[q] + b0 - m); z += e[q]; }
    const float inv = 1.f / z;
    for (int q = 0; q < QN; ++q) sFw[q] = e[q] * inv;
  }
  __syncthreads();

  // out[n,d] = sum_q fw[q] * pt[q,d]  (two half-waves contribute per column)
#pragma unroll
  for (int t = 0; t < 4; ++t) {
    float s = 0.f;
#pragma unroll
    for (int j = 0; j < 8; ++j) s += sFw[j + hi * 8] * ptv[t][j];
    atomicAdd(&sOut[c0 + t * 16 + lm], s);
  }
  __syncthreads();
  for (int i = tid; i < DIM; i += 256) out[(size_t)n * DIM + i] = sOut[i];
}

// ---------------------------------------------------------------------------
extern "C" void kernel_launch(void* const* d_in, const int* in_sizes, int n_in,
                              void* d_out, int out_size, void* d_ws, size_t ws_size,
                              hipStream_t stream) {
  const float* patch = (const float*)d_in[0];
  const float* query = (const float*)d_in[1];   // [1,16,512] == Q0
  const float* Wk    = (const float*)d_in[2];
  const float* bk    = (const float*)d_in[3];
  const float* Wv    = (const float*)d_in[4];
  const float* bv    = (const float*)d_in[5];
  const float* Wf    = (const float*)d_in[6];
  const float* bfp   = (const float*)d_in[7];
  const int*   idx   = (const int*)d_in[8];
  const unsigned char* mask = (const unsigned char*)d_in[9];
  float* out = (float*)d_out;

  // workspace layout (float offsets; bf16 regions 32B aligned)
  float* ws_f    = (float*)d_ws;
  float*  VQ     = ws_f;                        // 16*512 f32
  float*  sself  = ws_f + 8192;                 // 16
  float*  bkq    = ws_f + 8208;                 // 16
  float*  spatch = ws_f + 8448;                 // 4096*16 f32
  bf16_t* A_t    = (bf16_t*)(ws_f + 74240);     // 16*512 bf16 (transposed)
  bf16_t* Wv_t   = (bf16_t*)(ws_f + 78336);     // 512*512 bf16 (transposed)
  bf16_t* Vp     = (bf16_t*)(ws_f + 209408);    // 4096*512 bf16

  k_proj_q<<<32,   256, 0, stream>>>(Wk, query, A_t);
  k_vq    <<<32,   256, 0, stream>>>(Wv, query, bv, VQ);
  k_self  <<<16,   256, 0, stream>>>(Wk, query, bk, sself, bkq);
  k_cvt_wv<<<1024, 256, 0, stream>>>(Wv, Wv_t);
  k_project<<<BL / 16, 256, 0, stream>>>(patch, Wv_t, bv, A_t, bkq, Vp, spatch);
  k_aggregate<<<NG, 256, 0, stream>>>(idx, mask, spatch, sself, VQ, Wf, bfp, Vp, out);
}
